// HighBlock_20761871909043
// MI455X (gfx1250) — matrix operand order repaired and checked by reading, not compiled, not run on hardware
//
#include <hip/hip_runtime.h>
#include <hip/hip_bf16.h>
#include <math.h>

typedef _Float16 v16h __attribute__((ext_vector_type(16)));
typedef _Float16 h8   __attribute__((ext_vector_type(8)));
typedef _Float16 h4   __attribute__((ext_vector_type(4)));
typedef float    v8f  __attribute__((ext_vector_type(8)));
typedef float    f4   __attribute__((ext_vector_type(4)));

union HV  { v16h v; h8 h[2]; h4 q[4]; _Float16 e[16]; };
union H8U { h8 v; h4 q[2]; };

__device__ __forceinline__ float sigf(float x) { return 1.0f / (1.0f + expf(-x)); }

// ---------------------------------------------------------------------------
// WMMA GEMM:  C[M,N] = A(MxK) @ W(N,K)^T + bias1 + bias2
// AMODE 0: A row-major, lda=K.
// AMODE 1: A is x in (B=4, IN=256, T=256, P=88) layout; logical row
//          r = (b*88+p)*256 + t, logical col = i; elem = x[b,i,t,p].
// Block: 256 threads = 8 waves; block tile = 32 rows x 128 cols.
// Wave w owns cols [w*16, w*16+16); computes 2 row tiles sharing one B frag.
// Requires M%32==0, N%128==0, K%32==0.
// ---------------------------------------------------------------------------
template <int AMODE>
__global__ __launch_bounds__(256) void wmma_gemm(
    const float* __restrict__ A, const float* __restrict__ W,
    const float* __restrict__ bias1, const float* __restrict__ bias2,
    float* __restrict__ C, int M, int N, int K)
{
    __shared__ _Float16 lA[32 * 32];        // [m][k]
    __shared__ _Float16 lB[8][16 * 32];     // per-wave [n][k]

    const int tid  = threadIdx.x;
    const int wave = tid >> 5;
    const int lane = tid & 31;
    const bool hi  = (lane >= 16);
    const int l15  = lane & 15;
    const int rowBase = blockIdx.x * 32;
    const int colBase = blockIdx.y * 128 + wave * 16;

    // B-staging assignment (per wave): lane -> (row n = lane/2, kq = 16*(lane&1))
    const int bRow = lane >> 1;
    const int bKq  = (lane & 1) << 4;

    v8f acc0 = {}, acc1 = {};

    for (int k0 = 0; k0 < K; k0 += 32) {
        // ---- stage A tile (32x32): 1024 halves, 4 per thread, b64 store ----
        {
            int idx = tid << 2;             // multiple of 4 halves (8B aligned)
            int r = idx >> 5, c = idx & 31;
            int row = rowBase + r;
            if (AMODE == 0) {
                f4 v = *(const f4*)&A[(size_t)row * K + (k0 + c)];
                *(h4*)&lA[idx] = __builtin_convertvector(v, h4);
            } else {
                int t = row & 255;          // T = 256
                int bp = row >> 8;          // b*88 + p
                int p = bp % 88, b = bp / 88;
                h4 hv;
#pragma unroll
                for (int u = 0; u < 4; ++u) {
                    int i2 = k0 + c + u;    // input-channel index
                    hv[u] = (_Float16)A[(((size_t)b * 256 + i2) * 256 + t) * 88 + p];
                }
                *(h4*)&lA[idx] = hv;
            }
        }
        // ---- stage this wave's B tile (16n x 32k): 16 halves/lane ----
        {
            const float* wp = &W[(size_t)(colBase + bRow) * K + (k0 + bKq)];
            H8U p0, p1;
            p0.q[0] = __builtin_convertvector(*(const f4*)(wp + 0),  h4);
            p0.q[1] = __builtin_convertvector(*(const f4*)(wp + 4),  h4);
            p1.q[0] = __builtin_convertvector(*(const f4*)(wp + 8),  h4);
            p1.q[1] = __builtin_convertvector(*(const f4*)(wp + 12), h4);
            *(h8*)&lB[wave][bRow * 32 + bKq]     = p0.v;
            *(h8*)&lB[wave][bRow * 32 + bKq + 8] = p1.v;
        }
        __syncthreads();

        // ---- gather fragments per CDNA5 WMMA layout ----
        HV a0, a1, bf;
        {
            int aoff0 = l15 * 32 + (hi ? 8 : 0);          // row tile 0, m=l15
            int aoff1 = (16 + l15) * 32 + (hi ? 8 : 0);   // row tile 1, m=16+l15
            a0.h[0] = *(const h8*)&lA[aoff0];
            a0.h[1] = *(const h8*)&lA[aoff0 + 16];
            a1.h[0] = *(const h8*)&lA[aoff1];
            a1.h[1] = *(const h8*)&lA[aoff1 + 16];
            int boff = l15 * 32 + (hi ? 8 : 0);           // k = 8*hi + e (e<8), 16 + 8*hi + (e-8)
            bf.h[0] = *(const h8*)&lB[wave][boff];
            bf.h[1] = *(const h8*)&lB[wave][boff + 16];
        }
        acc0 = __builtin_amdgcn_wmma_f32_16x16x32_f16(
                   false, a0.v, false, bf.v, (short)0, acc0, false, false);
        acc1 = __builtin_amdgcn_wmma_f32_16x16x32_f16(
                   false, a1.v, false, bf.v, (short)0, acc1, false, false);
        __syncthreads();
    }

    const int col = colBase + l15;
    float bv = bias1 ? bias1[col] : 0.0f;
    if (bias2) bv += bias2[col];
#pragma unroll
    for (int r = 0; r < 8; ++r) {
        int row = rowBase + r + (hi ? 8 : 0);
        C[(size_t)row * N + col]        = acc0[r] + bv;
        C[(size_t)(row + 16) * N + col] = acc1[r] + bv;
    }
}

// ---------------------------------------------------------------------------
// LSTM recurrence for one (layer, direction).
//   xi : [M, T, 256] precomputed x@Wih^T + b_ih + b_hh, rows m*T+t
//   whh: [256, 64] fp32
//   out: [M, T, 128], this direction writes columns [colOff, colOff+64)
// Block = 256 threads (8 waves) owns 16 batch rows; loops all T steps.
// Per step: g = xi[:,t,:] + h @ whh^T via WMMA (16x256, K=64), then gates.
// Wave w computes n-tiles 2w, 2w+1; whh fragments pinned in registers.
// ---------------------------------------------------------------------------
__global__ __launch_bounds__(256) void lstm_rec(
    const float* __restrict__ xi, const float* __restrict__ whh,
    float* __restrict__ out, int T, int colOff, int dir)
{
    __shared__ _Float16 hbuf[16 * 64];      // h state, f16, [m][u]
    __shared__ float    gbuf[16 * 256];     // pre-activation gates

    const int tid  = threadIdx.x;
    const int wave = tid >> 5;
    const int lane = tid & 31;
    const bool hi  = (lane >= 16);
    const int l15  = lane & 15;
    const int mBase = blockIdx.x * 16;

    // Preload B fragments of whh^T (K=64 -> 2 chunks, 2 n-tiles per wave).
    HV bfrag[2][2];
#pragma unroll
    for (int tile = 0; tile < 2; ++tile) {
        int n = wave * 32 + tile * 16 + l15;
#pragma unroll
        for (int kc = 0; kc < 2; ++kc) {
            const float* wp = &whh[(size_t)n * 64 + kc * 32 + (hi ? 8 : 0)];
#pragma unroll
            for (int u = 0; u < 4; ++u)
                bfrag[tile][kc].q[u] =
                    __builtin_convertvector(*(const f4*)(wp + 4 * u + (u >= 2 ? 8 : 0)), h4);
        }
    }

    for (int i = tid; i < 16 * 64; i += 256) hbuf[i] = (_Float16)0.0f;
    float c0 = 0.f, c1 = 0.f, c2 = 0.f, c3 = 0.f;
    __syncthreads();

    for (int s = 0; s < T; ++s) {
        int t = dir ? (T - 1 - s) : s;

        // A fragments: h (16x64 f16) from LDS, two K-chunks.
        HV a0, a1;
        {
            int base = l15 * 64 + (hi ? 8 : 0);
            a0.h[0] = *(const h8*)&hbuf[base];
            a0.h[1] = *(const h8*)&hbuf[base + 16];
            a1.h[0] = *(const h8*)&hbuf[base + 32];
            a1.h[1] = *(const h8*)&hbuf[base + 48];
        }

#pragma unroll
        for (int tile = 0; tile < 2; ++tile) {
            int nb = wave * 32 + tile * 16;
            v8f acc;
#pragma unroll
            for (int r = 0; r < 8; ++r) {
                int m = mBase + r + (hi ? 8 : 0);
                acc[r] = xi[((size_t)m * T + t) * 256 + nb + l15];
            }
            acc = __builtin_amdgcn_wmma_f32_16x16x32_f16(
                      false, a0.v, false, bfrag[tile][0].v, (short)0, acc, false, false);
            acc = __builtin_amdgcn_wmma_f32_16x16x32_f16(
                      false, a1.v, false, bfrag[tile][1].v, (short)0, acc, false, false);
#pragma unroll
            for (int r = 0; r < 8; ++r)
                gbuf[(r + (hi ? 8 : 0)) * 256 + nb + l15] = acc[r];
        }
        __syncthreads();

        // Gates + state update; thread owns 4 fixed (m,u) cells.
#pragma unroll
        for (int q = 0; q < 4; ++q) {
            int idx = q * 256 + tid;
            int m = idx >> 6, u = idx & 63;
            float gi = gbuf[m * 256 + u];
            float gf = gbuf[m * 256 + 64 + u];
            float gg = gbuf[m * 256 + 128 + u];
            float go = gbuf[m * 256 + 192 + u];
            float cc = (q == 0) ? c0 : (q == 1) ? c1 : (q == 2) ? c2 : c3;
            cc = sigf(gf) * cc + sigf(gi) * tanhf(gg);
            float hh = sigf(go) * tanhf(cc);
            if (q == 0) c0 = cc; else if (q == 1) c1 = cc;
            else if (q == 2) c2 = cc; else c3 = cc;
            hbuf[m * 64 + u] = (_Float16)hh;
            out[((size_t)(mBase + m) * T + t) * 128 + colOff + u] = hh;
        }
        __syncthreads();
    }
}

// ---------------------------------------------------------------------------
// Neighborhood attention, heads=1, head_dim=128.
// One block per (sequence n, position i). qkv rows (n*L+i)*384: q|k|v.
// ---------------------------------------------------------------------------
__global__ __launch_bounds__(128) void na1d(
    const float* __restrict__ qkv, const float* __restrict__ rpb,
    float* __restrict__ out, int L, int KW)
{
    __shared__ float sc[96];
    const int n = blockIdx.x / L;
    const int i = blockIdx.x % L;
    const int tid = threadIdx.x;
    const float scale = 0.08838834764831845f;   // 128^-0.5

    int start = i - KW / 2;
    if (start < 0) start = 0;
    if (start > L - KW) start = L - KW;

    const float* q = qkv + ((size_t)n * L + i) * 384;
    if (tid < KW) {
        int j = start + tid;
        const float* k = qkv + ((size_t)n * L + j) * 384 + 128;
        float dot = 0.f;
#pragma unroll 4
        for (int d = 0; d < 128; d += 4) {
            f4 qv = *(const f4*)&q[d];
            f4 kv = *(const f4*)&k[d];
            dot += qv.x * kv.x + qv.y * kv.y + qv.z * kv.z + qv.w * kv.w;
        }
        int rel = j - i + KW - 1;
        if (rel < 0) rel = 0;
        if (rel > 2 * KW - 2) rel = 2 * KW - 2;
        sc[tid] = dot * scale + rpb[rel];
    }
    __syncthreads();

    float mx = -1e30f;
    for (int j2 = 0; j2 < KW; ++j2) mx = fmaxf(mx, sc[j2]);
    float den = 0.f;
    for (int j2 = 0; j2 < KW; ++j2) den += expf(sc[j2] - mx);

    const int d = tid;
    float acc = 0.f;
    for (int j2 = 0; j2 < KW; ++j2) {
        float w = expf(sc[j2] - mx) / den;
        acc += w * qkv[((size_t)n * L + start + j2) * 384 + 256 + d];
    }
    out[((size_t)n * L + i) * 128 + d] = acc;
}

// (B*P, T, D) rows (b*88+p)*256+t  ->  (B*T, P, D) rows (b*256+t)*88+p
__global__ void permute_TP(const float* __restrict__ in, float* __restrict__ out,
                           long total)
{
    long idx = (long)blockIdx.x * blockDim.x + threadIdx.x;
    if (idx >= total) return;
    int  d  = (int)(idx & 127);
    long ro = idx >> 7;
    int  p  = (int)(ro % 88);
    long bt = ro / 88;
    int  t  = (int)(bt & 255);
    int  b  = (int)(bt >> 8);
    long in_idx = (((long)(b * 88 + p) * 256 + t) << 7) + d;
    out[idx] = in[in_idx];
}

// ---------------------------------------------------------------------------
extern "C" void kernel_launch(void* const* d_in, const int* in_sizes, int n_in,
                              void* d_out, int out_size, void* d_ws, size_t ws_size,
                              hipStream_t stream)
{
    (void)in_sizes; (void)n_in; (void)out_size; (void)ws_size;

    const float* x        = (const float*)d_in[0];
    const float* w_ih_l0  = (const float*)d_in[1];
    const float* w_hh_l0  = (const float*)d_in[2];
    const float* b_ih_l0  = (const float*)d_in[3];
    const float* b_hh_l0  = (const float*)d_in[4];
    const float* w_ih_l0r = (const float*)d_in[5];
    const float* w_hh_l0r = (const float*)d_in[6];
    const float* b_ih_l0r = (const float*)d_in[7];
    const float* b_hh_l0r = (const float*)d_in[8];
    const float* w_ih_l1  = (const float*)d_in[9];
    const float* w_hh_l1  = (const float*)d_in[10];
    const float* b_ih_l1  = (const float*)d_in[11];
    const float* b_hh_l1  = (const float*)d_in[12];
    const float* w_ih_l1r = (const float*)d_in[13];
    const float* w_hh_l1r = (const float*)d_in[14];
    const float* b_ih_l1r = (const float*)d_in[15];
    const float* b_hh_l1r = (const float*)d_in[16];
    const float* qkv_w_t  = (const float*)d_in[17];
    const float* qkv_b_t  = (const float*)d_in[18];
    const float* rpb_t    = (const float*)d_in[19];
    const float* proj_w_t = (const float*)d_in[20];
    const float* proj_b_t = (const float*)d_in[21];
    const float* qkv_w_f  = (const float*)d_in[22];
    const float* qkv_b_f  = (const float*)d_in[23];
    const float* rpb_f    = (const float*)d_in[24];
    const float* proj_w_f = (const float*)d_in[25];
    const float* proj_b_f = (const float*)d_in[26];

    // Dimensions: B=4, P=88, T=256, IN=256, Dm=128, H=64; M = B*P*T = 90112.
    const int M = 90112, T = 256;
    const int MT = M / 32;            // 2816 row tiles (32 rows/block)

    // Workspace layout (floats), lifetime-based reuse (~323 MB total):
    float* ws   = (float*)d_ws;
    float* XI   = ws;                         // 23,068,672 (M x 256)
    float* ATT  = ws;                         // reuses XI (dead by then)
    float* NAT  = ws + 11534336;              // reuses XI upper half
    float* OUT0 = ws + 23068672;              // 11,534,336 (M x 128)
    float* PERM = OUT0;                       // reuses OUT0
    float* OUT1 = ws + 34603008;              // 11,534,336
    float* QKV  = ws + 46137344;              // 34,603,008 (M x 384)

    dim3 blk(256);
    dim3 gN256(MT, 2), gN384(MT, 3), gN128(MT, 1);

    // ---- BiLSTM layer 0 (input = x via transposed gather, K=256) ----
    wmma_gemm<1><<<gN256, blk, 0, stream>>>(x, w_ih_l0,  b_ih_l0,  b_hh_l0,  XI, M, 256, 256);
    lstm_rec<<<22, blk, 0, stream>>>(XI, w_hh_l0,  OUT0, T, 0,  0);
    wmma_gemm<1><<<gN256, blk, 0, stream>>>(x, w_ih_l0r, b_ih_l0r, b_hh_l0r, XI, M, 256, 256);
    lstm_rec<<<22, blk, 0, stream>>>(XI, w_hh_l0r, OUT0, T, 64, 1);

    // ---- BiLSTM layer 1 (input = OUT0, K=128) ----
    wmma_gemm<0><<<gN256, blk, 0, stream>>>(OUT0, w_ih_l1,  b_ih_l1,  b_hh_l1,  XI, M, 256, 128);
    lstm_rec<<<22, blk, 0, stream>>>(XI, w_hh_l1,  OUT1, T, 0,  0);
    wmma_gemm<0><<<gN256, blk, 0, stream>>>(OUT0, w_ih_l1r, b_ih_l1r, b_hh_l1r, XI, M, 256, 128);
    lstm_rec<<<22, blk, 0, stream>>>(XI, w_hh_l1r, OUT1, T, 64, 1);

    // ---- NA1D over T (L=256, kernel 7) ----
    wmma_gemm<0><<<gN384, blk, 0, stream>>>(OUT1, qkv_w_t, qkv_b_t, nullptr, QKV, M, 384, 128);
    na1d<<<dim3(M), dim3(128), 0, stream>>>(QKV, rpb_t, ATT, 256, 7);
    wmma_gemm<0><<<gN128, blk, 0, stream>>>(ATT, proj_w_t, proj_b_t, nullptr, NAT, M, 128, 128);

    // ---- permute (B,P,T,D) -> (B,T,P,D) ----
    long tot = (long)M * 128;
    permute_TP<<<dim3((unsigned)((tot + 255) / 256)), blk, 0, stream>>>(NAT, PERM, tot);

    // ---- NA1D over P (L=88, kernel 87) ----
    wmma_gemm<0><<<gN384, blk, 0, stream>>>(PERM, qkv_w_f, qkv_b_f, nullptr, QKV, M, 384, 128);
    na1d<<<dim3(M), dim3(128), 0, stream>>>(QKV, rpb_f, ATT, 88, 87);
    wmma_gemm<0><<<gN128, blk, 0, stream>>>(ATT, proj_w_f, proj_b_f, nullptr,
                                            (float*)d_out, M, 128, 128);
}
